// MultiHeadedAttention_73349451481615
// MI455X (gfx1250) — matrix operand order repaired and checked by reading, not compiled, not run on hardware
//
#include <hip/hip_runtime.h>

#define E_DIM 1024
#define NHEAD 16
#define HDIM  64
#define SEQ   2048
#define NBATCH 2
#define MTOT (NBATCH * SEQ)   // 4096 tokens

typedef __attribute__((ext_vector_type(16))) __bf16       v16bf;
typedef __attribute__((ext_vector_type(8)))  float        v8f;
typedef int v4i __attribute__((vector_size(16)));
typedef __attribute__((address_space(1))) v4i* gas_v4i_ptr;
typedef __attribute__((address_space(3))) v4i* las_v4i_ptr;

struct U32B { uint4 lo, hi; };   // 32 bytes -> bit_cast to v16bf

// float -> bf16 round-to-nearest-even
__device__ __forceinline__ unsigned short f2bf(float f) {
  unsigned int u = __builtin_bit_cast(unsigned int, f);
  u = (u + 0x7fffu + ((u >> 16) & 1u)) >> 16;
  return (unsigned short)u;
}
__device__ __forceinline__ unsigned int pack2(float lo, float hi) {
  return (unsigned int)f2bf(lo) | ((unsigned int)f2bf(hi) << 16);
}

// ---- CDNA5 async global->LDS copy (ASYNCcnt path), with safe fallback ------
#if __has_builtin(__builtin_amdgcn_global_load_async_to_lds_b128)
#define HAVE_ASYNC 1
#else
#define HAVE_ASYNC 0
#endif

__device__ __forceinline__ void copy16_g2l(const void* g, void* l) {
#if HAVE_ASYNC
  __builtin_amdgcn_global_load_async_to_lds_b128(
      (gas_v4i_ptr)(unsigned long long)g,
      (las_v4i_ptr)(unsigned int)(unsigned long long)l,
      0, 0);
#else
  *(uint4*)l = *(const uint4*)g;
#endif
}
__device__ __forceinline__ void async_wait0() {
#if HAVE_ASYNC
#if __has_builtin(__builtin_amdgcn_s_wait_asynccnt)
  __builtin_amdgcn_s_wait_asynccnt(0);
#else
  asm volatile("s_wait_asynccnt 0" ::: "memory");
#endif
#endif
}

// ---------------------------------------------------------------------------
// Kernel 0a: one-time f32 -> bf16 conversion (x and the four weight matrices).
// ---------------------------------------------------------------------------
__global__ __launch_bounds__(256)
void cvt_bf16_kernel(const float* __restrict__ src,
                     unsigned short* __restrict__ dst, int n4) {
  int i = blockIdx.x * 256 + threadIdx.x;       // float4 units
  if (i < n4) {
    float4 v = *(const float4*)&src[(size_t)i * 4];
    uint2 p;
    p.x = pack2(v.x, v.y);
    p.y = pack2(v.z, v.w);
    *(uint2*)&dst[(size_t)i * 4] = p;
  }
}

// ---------------------------------------------------------------------------
// Kernel 0b: RoPE (cos, sin) table, [SEQ][32] float2.
// ---------------------------------------------------------------------------
__global__ __launch_bounds__(256)
void rope_table_kernel(float2* __restrict__ tab) {
  int i = blockIdx.x * 256 + threadIdx.x;       // 2048 * 32 entries
  if (i < SEQ * 32) {
    int s = i >> 5, idx = i & 31;
    float inv = __powf(10000.0f, -(float)(2 * idx) / 64.0f);
    float c, sn;
    __sincosf((float)s * inv, &sn, &c);
    tab[i] = make_float2(c, sn);
  }
}

// ---------------------------------------------------------------------------
// Kernel 1: fused QKV projection + RoPE, bf16 WMMA, double-buffered async LDS.
// grid = (MTOT/128, 48): y>>4 selects Q/K/V, y&15 the head. 8 waves/block.
// ---------------------------------------------------------------------------
__global__ __launch_bounds__(256)
void qkv_rope_kernel(const unsigned short* __restrict__ xbf,
                     const unsigned short* __restrict__ wqkv,  // [3][E][E] bf16
                     const float2* __restrict__ rope,          // [SEQ][32]
                     unsigned short* __restrict__ qb,
                     unsigned short* __restrict__ kb,
                     unsigned short* __restrict__ vb) {
  const int m0  = blockIdx.x * 128;
  const int mat = blockIdx.y >> 4;              // 0=Q 1=K 2=V
  const int h   = blockIdx.y & 15;
  const unsigned short* W = wqkv + (size_t)mat * E_DIM * E_DIM;
  unsigned short* out = (mat == 0) ? qb : (mat == 1) ? kb : vb;

  __shared__ unsigned short sB[2][64][40];      // double-buffered W tile

  const int t    = threadIdx.x;
  const int lane = t & 31, w = t >> 5;
  const int hn = lane & 15, hi = lane >> 4;
  const int arow = m0 + w * 16 + hn;            // A-layout: lane = M row
  const int sr = t >> 2, scol = (t & 3) * 8;    // this thread's stage chunk

  v8f acc[4] = {};

  // prologue: fill buffer 0
  copy16_g2l(&W[(size_t)(h * HDIM + sr) * E_DIM + scol], &sB[0][sr][scol]);
  async_wait0();
  __syncthreads();

  for (int k0 = 0; k0 < E_DIM; k0 += 32) {
    const int cur = (k0 >> 5) & 1;
    if (k0 + 32 < E_DIM)                        // issue next tile, no wait yet
      copy16_g2l(&W[(size_t)(h * HDIM + sr) * E_DIM + k0 + 32 + scol],
                 &sB[cur ^ 1][sr][scol]);

    // A fragment: two contiguous 16B runs straight from bf16 global
    U32B au;
    au.lo = *(const uint4*)&xbf[(size_t)arow * E_DIM + k0 + 8 * hi];
    au.hi = *(const uint4*)&xbf[(size_t)arow * E_DIM + k0 + 16 + 8 * hi];
    if (k0 + 32 < E_DIM)
      __builtin_prefetch(&xbf[(size_t)arow * E_DIM + k0 + 32 + 8 * hi], 0, 3);
    v16bf a = __builtin_bit_cast(v16bf, au);

    #pragma unroll
    for (int f = 0; f < 4; ++f) {
      U32B bu;                                  // B 32x16: lane=N, lane[4]=K half
      bu.lo = *(const uint4*)&sB[cur][f * 16 + hn][8 * hi];
      bu.hi = *(const uint4*)&sB[cur][f * 16 + hn][16 + 8 * hi];
      v16bf b = __builtin_bit_cast(v16bf, bu);
      acc[f] = __builtin_amdgcn_wmma_f32_16x16x32_bf16(false, a, false, b,
                                                       (short)0, acc[f],
                                                       false, false);
    }
    async_wait0();        // next tile landed
    __syncthreads();      // everyone done with cur, next buffer valid
  }

  float racc[4][8];
  #pragma unroll
  for (int f = 0; f < 4; ++f)
    #pragma unroll
    for (int r = 0; r < 8; ++r) racc[f][r] = acc[f][r];

  // RoPE in-register via table: col d = 16f + hn; partner d+32 is fragment
  // f+2, same lane / same accumulator row; angle index = d & 31.
  if (mat < 2) {
    #pragma unroll
    for (int r = 0; r < 8; ++r) {
      int token = m0 + w * 16 + r + 8 * hi;
      int sidx = token & (SEQ - 1);
      float2 cs0 = rope[(size_t)sidx * 32 + hn];
      float2 cs1 = rope[(size_t)sidx * 32 + hn + 16];
      float x0 = racc[0][r], x1 = racc[1][r];
      float x2 = racc[2][r], x3 = racc[3][r];
      racc[0][r] = x0 * cs0.x - x2 * cs0.y;
      racc[2][r] = x2 * cs0.x + x0 * cs0.y;
      racc[1][r] = x1 * cs1.x - x3 * cs1.y;
      racc[3][r] = x3 * cs1.x + x1 * cs1.y;
    }
  }

  // store bf16 in [B, H, S, D]
  #pragma unroll
  for (int f = 0; f < 4; ++f)
    #pragma unroll
    for (int r = 0; r < 8; ++r) {
      int token = m0 + w * 16 + r + 8 * hi;
      int b = token >> 11, sidx = token & (SEQ - 1);
      out[(((size_t)(b * NHEAD + h)) * SEQ + sidx) * HDIM + f * 16 + hn] =
          f2bf(racc[f][r]);
    }
}

// ---------------------------------------------------------------------------
// Kernel 2: flash attention, double-buffered K/V tiles.
// grid = (SEQ/128, B*H), 256 threads (8 waves).
// ---------------------------------------------------------------------------
__global__ __launch_bounds__(256)
void flash_attn_kernel(const unsigned short* __restrict__ qb,
                       const unsigned short* __restrict__ kb,
                       const unsigned short* __restrict__ vb,
                       unsigned short* __restrict__ ctxb) {
  const int qt  = blockIdx.x;
  const int bh  = blockIdx.y;
  const int qlo = qt * 128;
  const int b = bh >> 4, h = bh & 15;

  __shared__ unsigned short sK [2][64][72];    // K block row-major [key][d]
  __shared__ unsigned short sVt[2][64][72];    // V block transposed [d][key]
  __shared__ unsigned short sP [8][16][72];    // per-wave P tile (C->A swizzle)

  const int t    = threadIdx.x;
  const int lane = t & 31, w = t >> 5;
  const int hn = lane & 15, hi = lane >> 4;

  const size_t kvbase = (size_t)bh * SEQ * HDIM;

  auto stageKV = [&](int bufi, int j0) {
    #pragma unroll
    for (int it = 0; it < 2; ++it) {           // K: 512 x 16B async chunks
      int i = t + it * 256;
      int key = i >> 3, c = (i & 7) * 8;
      copy16_g2l(&kb[kvbase + (size_t)(j0 + key) * HDIM + c],
                 &sK[bufi][key][c]);
    }
    #pragma unroll
    for (int it = 0; it < 8; ++it) {           // V transposed (manual)
      int i = t + it * 256;
      int key = i >> 5, dp = (i & 31) * 2;
      unsigned int vv =
          *(const unsigned int*)&vb[kvbase + (size_t)(j0 + key) * HDIM + dp];
      sVt[bufi][dp][key]     = (unsigned short)(vv & 0xffffu);
      sVt[bufi][dp + 1][key] = (unsigned short)(vv >> 16);
    }
  };

  // Q fragments (two 16x32 k-steps) held in registers for the whole kernel
  const size_t qrow = kvbase + (size_t)(qlo + w * 16 + hn) * HDIM;
  U32B q0u, q1u;
  q0u.lo = *(const uint4*)&qb[qrow + 8 * hi];
  q0u.hi = *(const uint4*)&qb[qrow + 16 + 8 * hi];
  q1u.lo = *(const uint4*)&qb[qrow + 32 + 8 * hi];
  q1u.hi = *(const uint4*)&qb[qrow + 48 + 8 * hi];
  v16bf qa0 = __builtin_bit_cast(v16bf, q0u);
  v16bf qa1 = __builtin_bit_cast(v16bf, q1u);

  float m_i[8], l_i[8], acc[4][8];
  #pragma unroll
  for (int r = 0; r < 8; ++r) { m_i[r] = -1e30f; l_i[r] = 0.f; }
  #pragma unroll
  for (int f = 0; f < 4; ++f)
    #pragma unroll
    for (int r = 0; r < 8; ++r) acc[f][r] = 0.f;

  const int jend = qlo + 128;                  // causal upper bound
  stageKV(0, 0);
  async_wait0();
  __syncthreads();

  for (int j0 = 0; j0 < jend; j0 += 64) {
    const int cur = (j0 >> 6) & 1;
    if (j0 + 64 < jend) stageKV(cur ^ 1, j0 + 64);   // issue next, no wait

    // S = Q K^T : 4 key-fragments x 2 d-steps
    float sc[4][8];
    #pragma unroll
    for (int f = 0; f < 4; ++f) {
      v8f c = {};
      #pragma unroll
      for (int tt = 0; tt < 2; ++tt) {
        U32B bu;
        bu.lo = *(const uint4*)&sK[cur][f * 16 + hn][tt * 32 + 8 * hi];
        bu.hi = *(const uint4*)&sK[cur][f * 16 + hn][tt * 32 + 16 + 8 * hi];
        v16bf bfr = __builtin_bit_cast(v16bf, bu);
        c = __builtin_amdgcn_wmma_f32_16x16x32_bf16(false, (tt == 0) ? qa0 : qa1,
                                                    false, bfr, (short)0, c,
                                                    false, false);
      }
      #pragma unroll
      for (int r = 0; r < 8; ++r) sc[f][r] = c[r];
    }

    // scale, causal mask, online softmax (row lives across 16 lanes)
    #pragma unroll
    for (int r = 0; r < 8; ++r) {
      int q = qlo + w * 16 + r + 8 * hi;
      float rm = -1e30f;
      #pragma unroll
      for (int f = 0; f < 4; ++f) {
        int key = j0 + f * 16 + hn;
        float v = sc[f][r] * 0.125f;           // 1/sqrt(64)
        v = (key > q) ? -1e30f : v;
        sc[f][r] = v;
        rm = fmaxf(rm, v);
      }
      #pragma unroll
      for (int off = 1; off < 16; off <<= 1)
        rm = fmaxf(rm, __shfl_xor(rm, off, 32));
      float mn    = fmaxf(m_i[r], rm);
      float alpha = __expf(m_i[r] - mn);
      float rs = 0.f;
      #pragma unroll
      for (int f = 0; f < 4; ++f) {
        float p = __expf(sc[f][r] - mn);
        sc[f][r] = p;
        rs += p;
      }
      #pragma unroll
      for (int off = 1; off < 16; off <<= 1)
        rs += __shfl_xor(rs, off, 32);
      l_i[r] = l_i[r] * alpha + rs;
      m_i[r] = mn;
      #pragma unroll
      for (int f = 0; f < 4; ++f) acc[f][r] *= alpha;
    }

    // C-layout -> A-layout swizzle of P through per-wave LDS tile
    #pragma unroll
    for (int f = 0; f < 4; ++f)
      #pragma unroll
      for (int r = 0; r < 8; ++r)
        sP[w][r + 8 * hi][f * 16 + hn] = f2bf(sc[f][r]);
    asm volatile("s_wait_dscnt 0" ::: "memory");  // wave-local RAW on sP

    // ctx += P V : 2 key-steps x 4 d-fragments
    #pragma unroll
    for (int tt = 0; tt < 2; ++tt) {
      U32B au;
      au.lo = *(const uint4*)&sP[w][hn][tt * 32 + 8 * hi];
      au.hi = *(const uint4*)&sP[w][hn][tt * 32 + 16 + 8 * hi];
      v16bf pa = __builtin_bit_cast(v16bf, au);
      #pragma unroll
      for (int f = 0; f < 4; ++f) {
        U32B bu;
        bu.lo = *(const uint4*)&sVt[cur][f * 16 + hn][tt * 32 + 8 * hi];
        bu.hi = *(const uint4*)&sVt[cur][f * 16 + hn][tt * 32 + 16 + 8 * hi];
        v16bf bfr = __builtin_bit_cast(v16bf, bu);
        v8f c;
        #pragma unroll
        for (int r = 0; r < 8; ++r) c[r] = acc[f][r];
        c = __builtin_amdgcn_wmma_f32_16x16x32_bf16(false, pa, false, bfr,
                                                    (short)0, c, false, false);
        #pragma unroll
        for (int r = 0; r < 8; ++r) acc[f][r] = c[r];
      }
    }

    async_wait0();        // next K tile landed
    __syncthreads();      // all waves done with cur buffers
  }

  // normalize and store ctx as bf16 in [B*S][E], e = h*64 + d
  #pragma unroll
  for (int r = 0; r < 8; ++r) {
    int token = qlo + w * 16 + r + 8 * hi;
    float inv = 1.0f / l_i[r];
    size_t row = (size_t)(b * SEQ + token) * E_DIM + h * HDIM;
    #pragma unroll
    for (int f = 0; f < 4; ++f)
      ctxb[row + f * 16 + hn] = f2bf(acc[f][r] * inv);
  }
}

// ---------------------------------------------------------------------------
// Kernel 3: output projection ctx @ Wo^T -> f32, double-buffered async LDS.
// grid = (MTOT/128, E/64).
// ---------------------------------------------------------------------------
__global__ __launch_bounds__(256)
void out_proj_kernel(const unsigned short* __restrict__ ctxb,
                     const unsigned short* __restrict__ wobf,
                     float* __restrict__ out) {
  const int m0 = blockIdx.x * 128;
  const int n0 = blockIdx.y * 64;

  __shared__ unsigned short sB[2][64][40];

  const int t    = threadIdx.x;
  const int lane = t & 31, w = t >> 5;
  const int hn = lane & 15, hi = lane >> 4;
  const int arow = m0 + w * 16 + hn;
  const int sr = t >> 2, scol = (t & 3) * 8;

  v8f acc[4] = {};

  copy16_g2l(&wobf[(size_t)(n0 + sr) * E_DIM + scol], &sB[0][sr][scol]);
  async_wait0();
  __syncthreads();

  for (int k0 = 0; k0 < E_DIM; k0 += 32) {
    const int cur = (k0 >> 5) & 1;
    if (k0 + 32 < E_DIM)
      copy16_g2l(&wobf[(size_t)(n0 + sr) * E_DIM + k0 + 32 + scol],
                 &sB[cur ^ 1][sr][scol]);

    U32B au;
    au.lo = *(const uint4*)&ctxb[(size_t)arow * E_DIM + k0 + 8 * hi];
    au.hi = *(const uint4*)&ctxb[(size_t)arow * E_DIM + k0 + 16 + 8 * hi];
    if (k0 + 32 < E_DIM)
      __builtin_prefetch(&ctxb[(size_t)arow * E_DIM + k0 + 32 + 8 * hi], 0, 3);
    v16bf a = __builtin_bit_cast(v16bf, au);

    #pragma unroll
    for (int f = 0; f < 4; ++f) {
      U32B bu;
      bu.lo = *(const uint4*)&sB[cur][f * 16 + hn][8 * hi];
      bu.hi = *(const uint4*)&sB[cur][f * 16 + hn][16 + 8 * hi];
      v16bf b = __builtin_bit_cast(v16bf, bu);
      acc[f] = __builtin_amdgcn_wmma_f32_16x16x32_bf16(false, a, false, b,
                                                       (short)0, acc[f],
                                                       false, false);
    }
    async_wait0();
    __syncthreads();
  }

  #pragma unroll
  for (int f = 0; f < 4; ++f)
    #pragma unroll
    for (int r = 0; r < 8; ++r) {
      int m = m0 + w * 16 + r + 8 * hi;
      out[(size_t)m * E_DIM + n0 + f * 16 + hn] = acc[f][r];
    }
}

// ---------------------------------------------------------------------------
extern "C" void kernel_launch(void* const* d_in, const int* in_sizes, int n_in,
                              void* d_out, int out_size, void* d_ws, size_t ws_size,
                              hipStream_t stream) {
  const float* x  = (const float*)d_in[0];
  const float* Wq = (const float*)d_in[1];
  const float* Wk = (const float*)d_in[2];
  const float* Wv = (const float*)d_in[3];
  const float* Wo = (const float*)d_in[4];

  const size_t xe = (size_t)MTOT * E_DIM;         // 4M elems
  const size_t we = (size_t)E_DIM * E_DIM;        // 1M elems

  unsigned short* xbf  = (unsigned short*)d_ws;   // [4M]
  unsigned short* wqkv = xbf + xe;                // [3M] (Wq|Wk|Wv)
  unsigned short* wobf = wqkv + 3 * we;           // [1M]
  unsigned short* qb   = wobf + we;               // [4M]
  unsigned short* kb   = qb + xe;                 // [4M]
  unsigned short* vb   = kb + xe;                 // [4M]
  unsigned short* ctxb = vb + xe;                 // [4M]
  float2* rope         = (float2*)(ctxb + xe);    // [SEQ*32] (cos,sin)

  dim3 blk(256);
  cvt_bf16_kernel<<<dim3((int)(xe / 4 / 256)), blk, 0, stream>>>(x, xbf, (int)(xe / 4));
  cvt_bf16_kernel<<<dim3((int)(we / 4 / 256)), blk, 0, stream>>>(Wq, wqkv, (int)(we / 4));
  cvt_bf16_kernel<<<dim3((int)(we / 4 / 256)), blk, 0, stream>>>(Wk, wqkv + we, (int)(we / 4));
  cvt_bf16_kernel<<<dim3((int)(we / 4 / 256)), blk, 0, stream>>>(Wv, wqkv + 2 * we, (int)(we / 4));
  cvt_bf16_kernel<<<dim3((int)(we / 4 / 256)), blk, 0, stream>>>(Wo, wobf, (int)(we / 4));
  rope_table_kernel<<<dim3(SEQ * 32 / 256), blk, 0, stream>>>(rope);

  qkv_rope_kernel<<<dim3(MTOT / 128, 3 * NHEAD), blk, 0, stream>>>(
      xbf, wqkv, rope, qb, kb, vb);
  flash_attn_kernel<<<dim3(SEQ / 128, NBATCH * NHEAD), blk, 0, stream>>>(
      qb, kb, vb, ctxb);
  out_proj_kernel<<<dim3(MTOT / 128, E_DIM / HDIM), blk, 0, stream>>>(
      ctxb, wobf, (float*)d_out);
}
